// GCN_14336600834353
// MI455X (gfx1250) — matrix operand order repaired and checked by reading, not compiled, not run on hardware
//
#include <hip/hip_runtime.h>

// ---------------------------------------------------------------------------
// GCN 2-layer forward for MI455X (gfx1250), wave32.
//   h1  = relu( Agg( x @ W1^T ) + b1 )
//   out =       Agg( h1 @ W2^T ) + b2
// GEMMs: bf16 WMMA (v_wmma_f32_16x16x32_bf16), LDS double-buffered tiles,
// staged with gfx1250 async global->LDS copies (ASYNCcnt). Layer-1 fuses the
// f32->bf16 demotion of x into A-staging (x read exactly once from HBM).
// Edge scatter: f32 atomics into an L2-resident (51MB << 192MB L2) buffer.
// ---------------------------------------------------------------------------

typedef __attribute__((ext_vector_type(16))) __bf16 v16bf;
typedef __attribute__((ext_vector_type(8)))  float  v8f;

#define CH 256   // output channels (both layers)
#define KC 32    // K-chunk per pipeline stage (== WMMA K)
#define ST 40    // padded LDS row stride in elems (80B: 16B-aligned, no bank conflicts)
#define MT 32    // M rows per block

// ---- gfx1250 async global->LDS helpers (ASYNCcnt path) --------------------
__device__ __forceinline__ unsigned lds_off(const void* p) {
  // generic pointer to LDS: addr[31:0] is the LDS byte address (ISA 10.2)
  return (unsigned)(uintptr_t)p;
}
__device__ __forceinline__ void async_b128(const void* g, unsigned lds) {
  asm volatile("global_load_async_to_lds_b128 %0, %1, off"
               :: "v"(lds), "v"((unsigned long long)(uintptr_t)g) : "memory");
}
__device__ __forceinline__ void async_b64(const void* g, unsigned lds) {
  asm volatile("global_load_async_to_lds_b64 %0, %1, off"
               :: "v"(lds), "v"((unsigned long long)(uintptr_t)g) : "memory");
}
__device__ __forceinline__ void wait_async0() {
  asm volatile("s_wait_asynccnt 0x0" ::: "memory");
}

// ---------------- elementwise helpers ----------------

__global__ void k_fill_f32(float* __restrict__ p, float v, int n) {
  int i = blockIdx.x * blockDim.x + threadIdx.x;
  if (i < n) p[i] = v;
}

__global__ void k_cvt_bf16(const float* __restrict__ s, __bf16* __restrict__ d, int n) {
  int i = blockIdx.x * blockDim.x + threadIdx.x;
  if (i < n) d[i] = (__bf16)s[i];
}

__global__ void k_deg_count(const int* __restrict__ dst, float* __restrict__ deg, int E) {
  int e = blockIdx.x * blockDim.x + threadIdx.x;
  if (e < E) atomicAdd(&deg[dst[e]], 1.0f);
}

__global__ void k_rsqrt_inplace(float* __restrict__ p, int n) {
  int i = blockIdx.x * blockDim.x + threadIdx.x;
  if (i < n) p[i] = rsqrtf(p[i]);
}

// ---------------- WMMA GEMM:  C[M,256] = A[M,K] @ B[256,K](bf16)^T ---------
// 256 threads (8 waves). Block tile 32(M) x 256(N); wave w -> msub=(w&1)*16,
// n0=(w>>1)*64, i.e. a 16x64 subtile with 4 f32 accumulators.
// K pipelined in KC=32 chunks, double-buffered in LDS:
//   B tile 256xKC staged via global_load_async_to_lds_b128 (coalesced 64B/row)
//   A tile  MTxKC: AF32 -> f32 loads + in-register bf16 demotion + ds_store
//                  else -> global_load_async_to_lds_b64
// Fragments then come from LDS (ds_load_b128), in this wave32 layout (A/B in contiguous K order):
//   A elem j, lane L: M=L&15, K = j + 16*(L>=16)  -> 2x16B chunks
//   B elem j, lane L: N=L&15, K = j + 16*(L>=16)            -> 2x16B chunks
//   D vgpr r, lane L: M = r + 8*(L>=16), N = L&15
template <bool AF32>
__global__ __launch_bounds__(256) void k_gemm_wmma_lds(
    const void* __restrict__ Av, const __bf16* __restrict__ B,
    float* __restrict__ C, int M, int K)
{
  __shared__ __bf16 Ash[2][MT * ST];   //  2 x 2.5 KB
  __shared__ __bf16 Bsh[2][CH * ST];   //  2 x 20 KB

  const int t    = threadIdx.x;
  const int lane = t & 31;
  const int wave = t >> 5;
  const int lh   = lane >> 4;      // lane half
  const int l15  = lane & 15;
  const int msub = (wave & 1) << 4;
  const int n0   = (wave >> 1) << 6;
  const int m0   = blockIdx.x * MT;

  // staging coordinates
  const int ar = t >> 3;           // A row 0..31 (8 threads/row, 4 elems each)
  const int ac = (t & 7) << 2;     // A col {0,4,...,28}
  int agrow = m0 + ar; if (agrow >= M) agrow = M - 1;   // clamp tail reads
  const int br = t >> 2;           // B row 0..63 (+64p), 4 threads/row
  const int bc = (t & 3) << 3;     // B col {0,8,16,24}

  const float*  Af = (const float*) Av;
  const __bf16* Ab = (const __bf16*)Av;

  v8f acc0 = {}, acc1 = {}, acc2 = {}, acc3 = {};

  auto stage = [&](int buf, int kc) {
    // B tile: 256 x KC bf16 via async copy (4 x 16B per thread)
#pragma unroll
    for (int p = 0; p < 4; ++p) {
      const int row = br + (p << 6);
      async_b128(B + (size_t)row * K + kc + bc,
                 lds_off(&Bsh[buf][row * ST + bc]));
    }
    // A tile: MT x KC
    if constexpr (AF32) {
      float4 a4 = *(const float4*)(Af + (size_t)agrow * K + kc + ac);
      union { unsigned long long u; __bf16 h[4]; } pk;
      pk.h[0] = (__bf16)a4.x; pk.h[1] = (__bf16)a4.y;
      pk.h[2] = (__bf16)a4.z; pk.h[3] = (__bf16)a4.w;
      *(unsigned long long*)&Ash[buf][ar * ST + ac] = pk.u;
    } else {
      async_b64(Ab + (size_t)agrow * K + kc + ac,
                lds_off(&Ash[buf][ar * ST + ac]));
    }
  };

  stage(0, 0);
  wait_async0();
  __syncthreads();

  int buf = 0;
  for (int kc = 0; kc < K; kc += KC) {
    if (kc + KC < K) stage(buf ^ 1, kc + KC);   // overlap with compute below

    union { v16bf v; uint4 u[2]; } af, b0, b1, b2, b3;
    const __bf16* ap = &Ash[buf][(msub + l15) * ST + (lh << 4)];
    af.u[0] = *(const uint4*)(ap);
    af.u[1] = *(const uint4*)(ap + 8);
    const __bf16* bp = &Bsh[buf][(n0 + l15) * ST + (lh << 4)];
    b0.u[0] = *(const uint4*)(bp);            b0.u[1] = *(const uint4*)(bp + 8);
    b1.u[0] = *(const uint4*)(bp + 16 * ST);  b1.u[1] = *(const uint4*)(bp + 16 * ST + 8);
    b2.u[0] = *(const uint4*)(bp + 32 * ST);  b2.u[1] = *(const uint4*)(bp + 32 * ST + 8);
    b3.u[0] = *(const uint4*)(bp + 48 * ST);  b3.u[1] = *(const uint4*)(bp + 48 * ST + 8);

    acc0 = __builtin_amdgcn_wmma_f32_16x16x32_bf16(false, af.v, false, b0.v, (short)0, acc0, false, false);
    acc1 = __builtin_amdgcn_wmma_f32_16x16x32_bf16(false, af.v, false, b1.v, (short)0, acc1, false, false);
    acc2 = __builtin_amdgcn_wmma_f32_16x16x32_bf16(false, af.v, false, b2.v, (short)0, acc2, false, false);
    acc3 = __builtin_amdgcn_wmma_f32_16x16x32_bf16(false, af.v, false, b3.v, (short)0, acc3, false, false);

    wait_async0();      // staged data for next chunk is in LDS
    __syncthreads();
    buf ^= 1;
  }

  const int crow = m0 + msub + (lh << 3);
  float* cbase = C + (size_t)crow * CH + n0 + l15;
#pragma unroll
  for (int r = 0; r < 8; ++r) {
    if (crow + r < M) {
      cbase[(size_t)r * CH]      = acc0[r];
      cbase[(size_t)r * CH + 16] = acc1[r];
      cbase[(size_t)r * CH + 32] = acc2[r];
      cbase[(size_t)r * CH + 48] = acc3[r];
    }
  }
}

// ---------------- edge scatter: agg[dst] += h[src] * dinv[s]*dinv[d] -------
// One block per edge, 256 threads = 256 channels. h[src] read coalesced;
// f32 atomics land in the 192MB L2 (agg buffer is only 51MB).
__global__ __launch_bounds__(256) void k_scatter_edges(
    const float* __restrict__ h, const int* __restrict__ src,
    const int* __restrict__ dst, const float* __restrict__ dinv,
    float* __restrict__ agg)
{
  const int e = blockIdx.x;
  const int c = threadIdx.x;
  const int s = src[e];
  const int d = dst[e];
  const float nrm = dinv[s] * dinv[d];
  atomicAdd(&agg[(size_t)d * CH + c], h[(size_t)s * CH + c] * nrm);
}

// ---------------- epilogues ------------------------------------------------
// layer1: h1_bf16 = bf16( relu( agg + h*dinv^2 + b ) )
__global__ void k_epilogue_relu_bf16(
    const float* __restrict__ agg, const float* __restrict__ lin,
    const float* __restrict__ dinv, const float* __restrict__ bias,
    __bf16* __restrict__ out, int total)
{
  int i = blockIdx.x * blockDim.x + threadIdx.x;
  if (i < total) {
    int n = i >> 8, c = i & 255;
    float di = dinv[n];
    float v = agg[i] + lin[i] * di * di + bias[c];
    out[i] = (__bf16)fmaxf(v, 0.0f);
  }
}

// layer2 (in place on d_out, which already holds the scattered agg):
__global__ void k_epilogue_out(
    float* __restrict__ outagg, const float* __restrict__ lin,
    const float* __restrict__ dinv, const float* __restrict__ bias, int total)
{
  int i = blockIdx.x * blockDim.x + threadIdx.x;
  if (i < total) {
    int n = i >> 8, c = i & 255;
    float di = dinv[n];
    outagg[i] = outagg[i] + lin[i] * di * di + bias[c];
  }
}

// ---------------------------------------------------------------------------

extern "C" void kernel_launch(void* const* d_in, const int* in_sizes, int n_in,
                              void* d_out, int out_size, void* d_ws, size_t ws_size,
                              hipStream_t stream)
{
  const float* x  = (const float*)d_in[0];   // [N, 512]
  const int*   ei = (const int*)  d_in[1];   // [2, E]
  const float* W1 = (const float*)d_in[2];   // [256, 512]
  const float* b1 = (const float*)d_in[3];   // [256]
  const float* W2 = (const float*)d_in[4];   // [256, 256]
  const float* b2 = (const float*)d_in[5];   // [256]
  float* out = (float*)d_out;                // [N, 256]

  const int NIN = in_sizes[2] / CH;          // 512
  const int N   = in_sizes[0] / NIN;         // 50000
  const int E   = in_sizes[1] / 2;           // 400000
  const int* srcI = ei;
  const int* dstI = ei + E;

  // workspace carve-out (256B aligned)
  char* ws = (char*)d_ws;
  size_t off = 0;
  auto carve = [&](size_t bytes) -> void* {
    void* p = ws + off;
    off = (off + bytes + 255) & ~(size_t)255;
    return p;
  };
  float*  dinv = (float*) carve((size_t)N * 4);
  __bf16* W1bf = (__bf16*)carve((size_t)CH * NIN * 2);
  __bf16* W2bf = (__bf16*)carve((size_t)CH * CH * 2);
  float*  lin  = (float*) carve((size_t)N * CH * 4);   // GEMM out (both layers)
  __bf16* h1bf = (__bf16*)carve((size_t)N * CH * 2);
  float*  agg  = (float*) carve((size_t)N * CH * 4);   // layer-1 scatter target

  const int T = 256;
  const int NC = N * CH;        // 12.8M

  // --- degrees: deg = 1 + in-degree; dinv = rsqrt(deg)
  k_fill_f32      <<<(N + T - 1) / T, T, 0, stream>>>(dinv, 1.0f, N);
  k_deg_count     <<<(E + T - 1) / T, T, 0, stream>>>(dstI, dinv, E);
  k_rsqrt_inplace <<<(N + T - 1) / T, T, 0, stream>>>(dinv, N);

  // --- weight demotion (tiny); x is demoted in-GEMM
  k_cvt_bf16 <<<(CH * NIN + T - 1) / T, T, 0, stream>>>(W1, W1bf, CH * NIN);
  k_cvt_bf16 <<<(CH * CH  + T - 1) / T, T, 0, stream>>>(W2, W2bf, CH * CH);

  // ================= layer 1 =================
  k_gemm_wmma_lds<true>  <<<(N + MT - 1) / MT, 256, 0, stream>>>(x, W1bf, lin, N, NIN);
  k_fill_f32             <<<(NC + T - 1) / T, T, 0, stream>>>(agg, 0.0f, NC);
  k_scatter_edges        <<<E, 256, 0, stream>>>(lin, srcI, dstI, dinv, agg);
  k_epilogue_relu_bf16   <<<(NC + T - 1) / T, T, 0, stream>>>(agg, lin, dinv, b1, h1bf, NC);

  // ================= layer 2 =================
  k_gemm_wmma_lds<false> <<<(N + MT - 1) / MT, 256, 0, stream>>>(h1bf, W2bf, lin, N, CH);
  k_fill_f32             <<<(NC + T - 1) / T, T, 0, stream>>>(out, 0.0f, NC);
  k_scatter_edges        <<<E, 256, 0, stream>>>(lin, srcI, dstI, dinv, out);
  k_epilogue_out         <<<(NC + T - 1) / T, T, 0, stream>>>(out, lin, dinv, b2, NC);
}
